// Head_34248069218900
// MI455X (gfx1250) — matrix operand order repaired and checked by reading, not compiled, not run on hardware
//
#include <hip/hip_runtime.h>

// Problem constants (match reference)
#define SEQT   256
#define EMB    384
#define HEADD  64
#define NWAVES 16
#define BLOCKSZ 512

// LDS row strides in halfs: every row start 16B-aligned, padded vs bank conflicts
#define WS 392   // weight rows (384+8):   784B = 16*49
#define XS 48    // x-chunk rows (32+16):   96B = 16*6
#define QS 72    // q/k rows (64+8):       144B = 16*9
#define VS 264   // v^T rows (256+8):      528B = 16*33
#define PSTR 48  // P scratch rows (32+16)

typedef __attribute__((ext_vector_type(16))) _Float16 v16h;
typedef __attribute__((ext_vector_type(8)))  _Float16 v8h;
typedef __attribute__((ext_vector_type(4)))  _Float16 v4h;
typedef __attribute__((ext_vector_type(8)))  float    v8f;

union V16 { v16h v; struct { v8h lo; v8h hi; } h; };

__device__ __forceinline__ v8f wmma16(const V16& a, const V16& b, v8f c) {
  // D = A(16x32 f16) * B(32x16 f16) + C(16x16 f32)
  return __builtin_amdgcn_wmma_f32_16x16x32_f16(false, a.v, false, b.v,
                                                (short)0, c, false, false);
}

// reductions across the 16 lanes of each wave32 half (xor masks <16 stay in-half)
__device__ __forceinline__ float halfmax16(float v) {
  v = fmaxf(v, __shfl_xor(v, 1));
  v = fmaxf(v, __shfl_xor(v, 2));
  v = fmaxf(v, __shfl_xor(v, 4));
  v = fmaxf(v, __shfl_xor(v, 8));
  return v;
}
__device__ __forceinline__ float halfsum16(float v) {
  v += __shfl_xor(v, 1);
  v += __shfl_xor(v, 2);
  v += __shfl_xor(v, 4);
  v += __shfl_xor(v, 8);
  return v;
}

__global__ __launch_bounds__(BLOCKSZ) void attn_head_fused(
    const float* __restrict__ x, const float* __restrict__ Wq,
    const float* __restrict__ Wk, const float* __restrict__ Wv,
    float* __restrict__ out)
{
  // 282,624 bytes total (<= 320KB WGP LDS)
  __shared__ _Float16 s_wT[3 * HEADD * WS];  // weights^T f16; reused as P scratch in phase 2
  __shared__ _Float16 s_x [SEQT * XS];       // streamed x chunk [256 x 32] f16
  __shared__ _Float16 s_Q [SEQT * QS];       // q [t][h] f16
  __shared__ _Float16 s_K [SEQT * QS];       // k [s][h] f16
  __shared__ _Float16 s_Vt[HEADD * VS];      // v^T [h][s] f16

  const int tid  = threadIdx.x;
  const int lane = tid & 31;
  const int wid  = tid >> 5;     // 0..15
  const int bb   = blockIdx.x;   // batch
  const int nloc = lane & 15;    // C/D: column; A: row; B: column
  const int hi   = lane >> 4;
  const int koffA = hi ? 16 : 0; // A-matrix 16x32 f16 lane K base (contiguous, matches B)
  const int koffB = hi ? 16 : 0; // B-matrix 32x16 f16 lane K base

  const float* xb = x + (size_t)bb * SEQT * EMB;

  // ---------------- stage weights: wT[p][n][k] = W_p[k][n] as f16 ----------------
  {
    const float* Wsrc[3] = {Wq, Wk, Wv};
    for (int idx = tid; idx < 3 * EMB * HEADD; idx += BLOCKSZ) {
      int pj  = idx / (EMB * HEADD);
      int rem = idx - pj * (EMB * HEADD);
      int k = rem >> 6, n = rem & 63;
      s_wT[(pj * HEADD + n) * WS + k] = (_Float16)Wsrc[pj][k * HEADD + n];
    }
  }

  // ---------------- phase 1: q,k,v = x @ {Wq,Wk,Wv}  (wave wid owns 16 rows) -----
  v8f acc[12];  // 3 projections x 4 n-tiles, 16x16 f32 each
#pragma unroll
  for (int i = 0; i < 12; ++i) acc[i] = (v8f){0,0,0,0,0,0,0,0};

  const int arow = wid * 16 + nloc;  // this lane's A row (global t)

  for (int kc = 0; kc < 12; ++kc) {
    __syncthreads();
    // cooperative stage of x[:, kc*32 : kc*32+32] -> f16, coalesced float4 loads
#pragma unroll
    for (int it = 0; it < 4; ++it) {
      int g   = it * BLOCKSZ + tid;      // 0..2047
      int row = g >> 3;
      int c4  = (g & 7) * 4;
      float4 f = *reinterpret_cast<const float4*>(xb + row * EMB + kc * 32 + c4);
      v4h hq;
      hq[0] = (_Float16)f.x; hq[1] = (_Float16)f.y;
      hq[2] = (_Float16)f.z; hq[3] = (_Float16)f.w;
      *reinterpret_cast<v4h*>(&s_x[row * XS + c4]) = hq;
    }
    __syncthreads();

    V16 a;
    a.h.lo = *reinterpret_cast<const v8h*>(&s_x[arow * XS + koffA]);
    a.h.hi = *reinterpret_cast<const v8h*>(&s_x[arow * XS + koffA + 8]);
#pragma unroll
    for (int pj = 0; pj < 3; ++pj) {
#pragma unroll
      for (int nt = 0; nt < 4; ++nt) {
        int n = nt * 16 + nloc;
        const _Float16* wrow = &s_wT[(pj * HEADD + n) * WS + kc * 32 + koffB];
        V16 bm;
        bm.h.lo = *reinterpret_cast<const v8h*>(wrow);
        bm.h.hi = *reinterpret_cast<const v8h*>(wrow + 8);
        acc[pj * 4 + nt] = wmma16(a, bm, acc[pj * 4 + nt]);
      }
    }
  }

  // scatter projections to LDS (C-layout: lane col nloc, VGPR r -> row r+8*hi)
#pragma unroll
  for (int pj = 0; pj < 3; ++pj) {
#pragma unroll
    for (int nt = 0; nt < 4; ++nt) {
#pragma unroll
      for (int r = 0; r < 8; ++r) {
        int m = wid * 16 + r + 8 * hi;
        int n = nt * 16 + nloc;
        _Float16 hv = (_Float16)acc[pj * 4 + nt][r];
        if      (pj == 0) s_Q [m * QS + n] = hv;
        else if (pj == 1) s_K [m * QS + n] = hv;
        else              s_Vt[n * VS + m] = hv;   // transposed for PV B-operand
      }
    }
  }
  __syncthreads();

  // ---------------- phase 2: causal flash attention (wave wid owns q-tile wid) ---
  _Float16* pS = s_wT + wid * (16 * PSTR);  // per-wave scratch, reuses dead weight LDS
  const int qt = wid;
  const int q0 = qt * 16;
  const float scale = 0.05103103630798288f;   // 384^-0.5 (reference scales by C^-0.5)
  const float L2E   = 1.4426950408889634f;

  V16 aQ0, aQ1;  // Q A-operands for h=0..31 and h=32..63
  {
    int qr = q0 + nloc;
    aQ0.h.lo = *reinterpret_cast<const v8h*>(&s_Q[qr * QS + koffA]);
    aQ0.h.hi = *reinterpret_cast<const v8h*>(&s_Q[qr * QS + koffA + 8]);
    aQ1.h.lo = *reinterpret_cast<const v8h*>(&s_Q[qr * QS + 32 + koffA]);
    aQ1.h.hi = *reinterpret_cast<const v8h*>(&s_Q[qr * QS + 32 + koffA + 8]);
  }

  v8f o[4];
#pragma unroll
  for (int i = 0; i < 4; ++i) o[i] = (v8f){0,0,0,0,0,0,0,0};
  float mrow[8], lrow[8];
#pragma unroll
  for (int r = 0; r < 8; ++r) { mrow[r] = -3.0e38f; lrow[r] = 0.0f; }

  const int nblk = (qt >> 1) + 1;  // causal: 32-wide key blocks up to the diagonal
  for (int jj = 0; jj < nblk; ++jj) {
    int s0 = jj * 32;
    v8f sc0 = (v8f){0,0,0,0,0,0,0,0};
    v8f sc1 = (v8f){0,0,0,0,0,0,0,0};
    {  // S tile 0: key cols s0..s0+15
      int s = s0 + nloc;
      V16 b0, b1;
      b0.h.lo = *reinterpret_cast<const v8h*>(&s_K[s * QS + koffB]);
      b0.h.hi = *reinterpret_cast<const v8h*>(&s_K[s * QS + koffB + 8]);
      b1.h.lo = *reinterpret_cast<const v8h*>(&s_K[s * QS + 32 + koffB]);
      b1.h.hi = *reinterpret_cast<const v8h*>(&s_K[s * QS + 32 + koffB + 8]);
      sc0 = wmma16(aQ0, b0, sc0);
      sc0 = wmma16(aQ1, b1, sc0);
    }
    {  // S tile 1: key cols s0+16..s0+31
      int s = s0 + 16 + nloc;
      V16 b0, b1;
      b0.h.lo = *reinterpret_cast<const v8h*>(&s_K[s * QS + koffB]);
      b0.h.hi = *reinterpret_cast<const v8h*>(&s_K[s * QS + koffB + 8]);
      b1.h.lo = *reinterpret_cast<const v8h*>(&s_K[s * QS + 32 + koffB]);
      b1.h.hi = *reinterpret_cast<const v8h*>(&s_K[s * QS + 32 + koffB + 8]);
      sc1 = wmma16(aQ0, b0, sc1);
      sc1 = wmma16(aQ1, b1, sc1);
    }

    // online softmax update (fp32)
    float p0[8], p1[8], alpha[8];
#pragma unroll
    for (int r = 0; r < 8; ++r) {
      int q = q0 + r + 8 * hi;
      float v0 = sc0[r] * scale;
      float v1 = sc1[r] * scale;
      if (s0 + nloc      > q) v0 = -3.0e38f;  // causal mask
      if (s0 + 16 + nloc > q) v1 = -3.0e38f;
      float rm   = halfmax16(fmaxf(v0, v1));
      float mnew = fmaxf(mrow[r], rm);
      alpha[r] = exp2f((mrow[r] - mnew) * L2E);
      p0[r]    = exp2f((v0 - mnew) * L2E);
      p1[r]    = exp2f((v1 - mnew) * L2E);
      lrow[r]  = lrow[r] * alpha[r] + halfsum16(p0[r] + p1[r]);
      mrow[r]  = mnew;
    }
#pragma unroll
    for (int ht = 0; ht < 4; ++ht)
#pragma unroll
      for (int r = 0; r < 8; ++r) o[ht][r] *= alpha[r];

    // transpose P (C-layout -> A-layout) through per-wave LDS scratch
#pragma unroll
    for (int r = 0; r < 8; ++r) {
      int m = r + 8 * hi;
      pS[m * PSTR + nloc]      = (_Float16)p0[r];
      pS[m * PSTR + 16 + nloc] = (_Float16)p1[r];
    }
    V16 aP;  // same-wave DS ops are in-order: safe without barrier
    aP.h.lo = *reinterpret_cast<const v8h*>(&pS[nloc * PSTR + koffA]);
    aP.h.hi = *reinterpret_cast<const v8h*>(&pS[nloc * PSTR + koffA + 8]);

#pragma unroll
    for (int ht = 0; ht < 4; ++ht) {
      int h = ht * 16 + nloc;
      V16 bv;
      bv.h.lo = *reinterpret_cast<const v8h*>(&s_Vt[h * VS + s0 + koffB]);
      bv.h.hi = *reinterpret_cast<const v8h*>(&s_Vt[h * VS + s0 + koffB + 8]);
      o[ht] = wmma16(aP, bv, o[ht]);
    }
  }

  // epilogue: normalize and store fp32 output [b][t][h]
  float* ob = out + ((size_t)bb * SEQT + q0) * HEADD;
#pragma unroll
  for (int ht = 0; ht < 4; ++ht) {
#pragma unroll
    for (int r = 0; r < 8; ++r) {
      int m = r + 8 * hi;
      ob[m * HEADD + ht * 16 + nloc] = o[ht][r] / lrow[r];
    }
  }
}

extern "C" void kernel_launch(void* const* d_in, const int* in_sizes, int n_in,
                              void* d_out, int out_size, void* d_ws, size_t ws_size,
                              hipStream_t stream) {
  const float* x  = (const float*)d_in[0];
  const float* Wq = (const float*)d_in[1];
  const float* Wk = (const float*)d_in[2];
  const float* Wv = (const float*)d_in[3];
  float* out = (float*)d_out;
  const int B = in_sizes[0] / (SEQT * EMB);  // 512
  attn_head_fused<<<B, BLOCKSZ, 0, stream>>>(x, Wq, Wk, Wv, out);
}
